// DANetHead_2_33947421508385
// MI455X (gfx1250) — matrix operand order repaired and checked by reading, not compiled, not run on hardware
//
#include <hip/hip_runtime.h>
#include <hip/hip_bf16.h>

typedef __attribute__((ext_vector_type(16))) _Float16 v16h;
typedef __attribute__((ext_vector_type(8)))  _Float16 v8h;
typedef __attribute__((ext_vector_type(8)))  float    v8f;

#define WAVE 32

__device__ __forceinline__ v8f wmma_f16(v16h a, v16h b, v8f c) {
  return __builtin_amdgcn_wmma_f32_16x16x32_f16(false, a, false, b, (short)0, c,
                                                false, false);
}
__device__ __forceinline__ v8h ldv8(const _Float16* p) { return *(const v8h*)p; }
__device__ __forceinline__ v16h cat16(v8h lo, v8h hi) {
  return __builtin_shufflevector(lo, hi, 0, 1, 2, 3, 4, 5, 6, 7, 8, 9, 10, 11, 12, 13, 14, 15);
}
// A fragment: lane half h needs K = {16h..16h+15}
__device__ __forceinline__ v16h load_a(const _Float16* row, int half) {
  return cat16(ldv8(row + 16 * half), ldv8(row + 16 * half + 8));
}
// B fragment: lane half h needs K = {16h..16h+15}
__device__ __forceinline__ v16h load_b(const _Float16* row, int half) {
  return cat16(ldv8(row + 16 * half), ldv8(row + 16 * half + 8));
}

// ---------------------------------------------------------------------------
// converts / repacks
// ---------------------------------------------------------------------------
__global__ void cvt_f32_f16(const float* __restrict__ in, _Float16* __restrict__ out, int n) {
  int i = blockIdx.x * blockDim.x + threadIdx.x;
  if (i < n) out[i] = (_Float16)in[i];
}

// x [B][C][H*W] f32 -> padded channel-last xp [B][(H+2)*(W+2)][C] f16, zero halo
__global__ void pad_chlast_f16(const float* __restrict__ in, _Float16* __restrict__ out,
                               int Bn, int C, int H, int W) {
  const int Wp = W + 2, Np = (H + 2) * Wp, N = H * W;
  int i = blockIdx.x * blockDim.x + threadIdx.x;
  if (i >= Bn * Np * C) return;
  int c = i % C, p = (i / C) % Np, b = i / (C * Np);
  int hp = p / Wp, wp = p % Wp;
  float v = 0.f;
  if (hp >= 1 && hp <= H && wp >= 1 && wp <= W)
    v = in[((size_t)b * C + c) * N + (hp - 1) * W + (wp - 1)];
  out[i] = (_Float16)v;
}

// zero only the halo cells of a padded channel-last buffer (interior written later)
__global__ void zero_border(_Float16* __restrict__ out, int Bn, int C, int H, int W) {
  const int Wp = W + 2, Np = (H + 2) * Wp;
  int i = blockIdx.x * blockDim.x + threadIdx.x;
  if (i >= Bn * Np * C) return;
  int p = (i / C) % Np;
  int hp = p / Wp, wp = p % Wp;
  if (hp == 0 || hp == H + 1 || wp == 0 || wp == W + 1) out[i] = (_Float16)0.0f;
}

// w [Co][Cin][3][3] f32 -> wt [tap][Co][Cin] f16
__global__ void repack_w3x3(const float* __restrict__ w, _Float16* __restrict__ wt,
                            int Co, int Cin) {
  int i = blockIdx.x * blockDim.x + threadIdx.x;
  if (i >= Co * Cin * 9) return;
  int tap = i % 9, ci = (i / 9) % Cin, co = i / (9 * Cin);
  wt[((size_t)tap * Co + co) * Cin + ci] = (_Float16)w[i];
}

// ---------------------------------------------------------------------------
// 3x3 SAME conv = 9 shifted 1x1 GEMMs on the padded layout (no masking).
// M2-blocked: one wave computes a 32(co) x 16(pos) tile, B fragment shared.
// xp: [B][(H+2)(W+2)][Cin] f16;  wt: [9][Co][Cin] f16;  y: [B][Co][H*W] f32
// ---------------------------------------------------------------------------
__global__ void conv3x3_wmma(const _Float16* __restrict__ xp,
                             const _Float16* __restrict__ wt,
                             float* __restrict__ y,
                             int Bn, int Cin, int Co, int H, int W) {
  const int N = H * W, Wp = W + 2, Np = (H + 2) * Wp;
  const int ntiles = N / 16, mtiles = Co / 32;
  int bid = blockIdx.x;
  int b  = bid / (ntiles * mtiles);
  int r  = bid % (ntiles * mtiles);
  int m0 = (r / ntiles) * 32;
  int n0 = (r % ntiles) * 16;
  const int lane = threadIdx.x & (WAVE - 1);
  const int half = lane >> 4, lm = lane & 15;
  const int n = n0 + lm;
  const int h0 = n / W, w0 = n % W;
  const _Float16* bbase = xp + ((size_t)b * Np + (h0 + 1) * Wp + (w0 + 1)) * Cin;

  v8f acc0 = {}, acc1 = {};
#pragma unroll
  for (int tap = 0; tap < 9; ++tap) {
    const int dy = tap / 3 - 1, dx = tap % 3 - 1;            // constants after unroll
    const _Float16* brow = bbase + (ptrdiff_t)(dy * Wp + dx) * Cin;
    const _Float16* arow = wt + ((size_t)tap * Co + m0 + lm) * Cin;
    __builtin_prefetch(arow + (size_t)Co * Cin, 0, 1);       // next tap's weights
    for (int c0 = 0; c0 < Cin; c0 += 32) {
      v16h bf = load_b(brow + c0, half);
      acc0 = wmma_f16(load_a(arow + c0, half), bf, acc0);
      acc1 = wmma_f16(load_a(arow + (size_t)16 * Cin + c0, half), bf, acc1);
    }
  }
#pragma unroll
  for (int rr = 0; rr < 8; ++rr) {
    int m = rr + 8 * half;
    y[((size_t)b * Co + m0 + m) * N + n0 + lm]      = acc0[rr];
    y[((size_t)b * Co + m0 + 16 + m) * N + n0 + lm] = acc1[rr];
  }
}

// ---------------------------------------------------------------------------
// Training-mode BN stats per channel -> scale/shift
// ---------------------------------------------------------------------------
__global__ void bn_stats(const float* __restrict__ y, const float* __restrict__ g,
                         const float* __restrict__ beta, float* __restrict__ scale,
                         float* __restrict__ shift, int Bn, int C, int N) {
  int c = blockIdx.x, tid = threadIdx.x;
  int total = Bn * N;
  float s = 0.f, s2 = 0.f;
  for (int i = tid; i < total; i += blockDim.x) {
    int b = i / N, n = i % N;
    float v = y[((size_t)b * C + c) * N + n];
    s += v; s2 += v * v;
  }
  __shared__ float sh0[256], sh1[256];
  sh0[tid] = s; sh1[tid] = s2; __syncthreads();
  for (int o = 128; o > 0; o >>= 1) {
    if (tid < o) { sh0[tid] += sh0[tid + o]; sh1[tid] += sh1[tid + o]; }
    __syncthreads();
  }
  if (tid == 0) {
    float mean = sh0[0] / total;
    float var  = sh1[0] / total - mean * mean;
    float sc   = g[c] * rsqrtf(var + 1e-5f);
    scale[c] = sc;
    shift[c] = beta[c] - mean * sc;
  }
}

// y (ch-first f32) -> relu(bn) into: outf (ch-first f32, may alias y),
// outh (ch-first f16, optional), outT (ch-last unpadded f16, optional)
__global__ void bn_relu_apply(const float* __restrict__ y, const float* __restrict__ scale,
                              const float* __restrict__ shift, float* __restrict__ outf,
                              _Float16* __restrict__ outh, _Float16* __restrict__ outT,
                              int C, int N, int total) {
  int i = blockIdx.x * blockDim.x + threadIdx.x;
  if (i >= total) return;
  int n = i % N, c = (i / N) % C, b = i / (C * N);
  float v = fmaxf(y[i] * scale[c] + shift[c], 0.f);
  outf[i] = v;
  if (outh) outh[i] = (_Float16)v;
  if (outT) outT[((size_t)b * N + n) * C + c] = (_Float16)v;
}

// ---------------------------------------------------------------------------
// q/k/v projections.  q/k channel-last zero-padded to 32: [B][N][32];
// v channel-first [B][Ci][N] (its WMMA K is the spatial dim).
// ---------------------------------------------------------------------------
__global__ void qkv_kernel(const float* __restrict__ feat1,
                           const float* __restrict__ wq, const float* __restrict__ bq,
                           const float* __restrict__ wk, const float* __restrict__ bk,
                           const float* __restrict__ wv, const float* __restrict__ bv,
                           _Float16* __restrict__ qT, _Float16* __restrict__ kT,
                           _Float16* __restrict__ vh, int Bn, int Ci, int Cq, int N) {
  int i = blockIdx.x * blockDim.x + threadIdx.x;
  if (i >= Bn * N) return;
  int b = i / N, n = i % N;
  const float* f = feat1 + (size_t)b * Ci * N + n;
  float fr[64];
  for (int c = 0; c < Ci; ++c) fr[c] = f[(size_t)c * N];
  for (int o = 0; o < Ci; ++o) {
    float acc = bv[o];
    for (int c = 0; c < Ci; ++c) acc += wv[o * Ci + c] * fr[c];
    vh[((size_t)b * Ci + o) * N + n] = (_Float16)acc;
  }
  _Float16* qrow = qT + ((size_t)b * N + n) * 32;
  _Float16* krow = kT + ((size_t)b * N + n) * 32;
  for (int o = 0; o < 32; ++o) {
    float aq = 0.f, ak = 0.f;
    if (o < Cq) {
      aq = bq[o]; ak = bk[o];
      for (int c = 0; c < Ci; ++c) { aq += wq[o * Ci + c] * fr[c]; ak += wk[o * Ci + c] * fr[c]; }
    }
    qrow[o] = (_Float16)aq;
    krow[o] = (_Float16)ak;
  }
}

// ---------------------------------------------------------------------------
// Fused PAM (flash-attention): one wave = 16 query positions; attn never in HBM.
// Writes sa_feat = gamma*pam + feat1 into the PADDED channel-last layout.
// ---------------------------------------------------------------------------
__global__ void pam_flash(const _Float16* __restrict__ qT, const _Float16* __restrict__ kT,
                          const _Float16* __restrict__ vh, const float* __restrict__ feat1,
                          const float* __restrict__ gamma, _Float16* __restrict__ saP,
                          int Bn, int Ci, int N, int H, int W) {
  __shared__ _Float16 pl[16][32];            // [i_local][j_local]
  __shared__ float mL[16], lL[16], fL[16];
  const int lane = threadIdx.x & (WAVE - 1);
  const int half = lane >> 4, lm = lane & 15;
  const int ntiles = N / 16;
  const int b  = blockIdx.x / ntiles;
  const int i0 = (blockIdx.x % ntiles) * 16;
  const float g = gamma[0];

  if (threadIdx.x < 16) { mL[threadIdx.x] = -3.0e38f; lL[threadIdx.x] = 0.f; }
  __syncthreads();

  v16h aq = load_a(qT + ((size_t)b * N + i0 + lm) * 32, half);

  v8f acc[4] = {{}, {}, {}, {}};             // D[m=c][n=i], 4 c-tiles (Ci=64)
  for (int j0 = 0; j0 < N; j0 += 32) {
    float e[2][8], rmax[8];
#pragma unroll
    for (int t = 0; t < 2; ++t) {
      v16h bkf = load_b(kT + ((size_t)b * N + j0 + t * 16 + lm) * 32, half);
      v8f ez = {};
      ez = wmma_f16(aq, bkf, ez);
#pragma unroll
      for (int r = 0; r < 8; ++r) e[t][r] = ez[r];
    }
#pragma unroll
    for (int r = 0; r < 8; ++r) {
      float v = fmaxf(e[0][r], e[1][r]);
      for (int o = 1; o < 16; o <<= 1) v = fmaxf(v, __shfl_xor(v, o, WAVE));
      rmax[r] = v;
    }
    __syncthreads();
    if (lm == 0) {
#pragma unroll
      for (int r = 0; r < 8; ++r) {
        int row = r + 8 * half;
        float mo = mL[row], mn = fmaxf(mo, rmax[r]);
        mL[row] = mn;
        fL[row] = __expf(mo - mn);
      }
    }
    __syncthreads();
    float rsum[8];
#pragma unroll
    for (int r = 0; r < 8; ++r) {
      int row = r + 8 * half;
      float mn = mL[row];
      float p0 = __expf(e[0][r] - mn), p1 = __expf(e[1][r] - mn);
      pl[row][lm]      = (_Float16)p0;
      pl[row][16 + lm] = (_Float16)p1;
      float rs = p0 + p1;
      for (int o = 1; o < 16; o <<= 1) rs += __shfl_xor(rs, o, WAVE);
      rsum[r] = rs;
    }
    __syncthreads();
    if (lm == 0) {
#pragma unroll
      for (int r = 0; r < 8; ++r) {
        int row = r + 8 * half;
        lL[row] = lL[row] * fL[row] + rsum[r];
      }
    }
    float fct = fL[lm];
    v16h bp = load_b((const _Float16*)&pl[lm][0], half);   // 2x ds_load_b128
#pragma unroll
    for (int ct = 0; ct < 4; ++ct) {
#pragma unroll
      for (int r = 0; r < 8; ++r) acc[ct][r] *= fct;
      v16h av = load_a(vh + ((size_t)b * Ci + ct * 16 + lm) * N + j0, half);
      acc[ct] = wmma_f16(av, bp, acc[ct]);
    }
  }
  __syncthreads();
  float invl = 1.f / lL[lm];
  const int Wp = W + 2, Np = (H + 2) * Wp;
  const int n = i0 + lm, h = n / W, w = n % W;
  const size_t nrow = ((size_t)b * Np + (h + 1) * Wp + (w + 1)) * Ci;
#pragma unroll
  for (int ct = 0; ct < 4; ++ct) {
    v8h hv;
#pragma unroll
    for (int r = 0; r < 8; ++r) {
      int c = ct * 16 + 8 * half + r;
      float fv = feat1[((size_t)b * Ci + c) * N + n];
      hv[r] = (_Float16)(g * (acc[ct][r] * invl) + fv);
    }
    *(v8h*)&saP[nrow + ct * 16 + 8 * half] = hv;           // contiguous-c store
  }
}

// ---------------------------------------------------------------------------
// CAM Gram: ce[b][c][d] = sum_n f[c][n] f[d][n], K = N (both frags contiguous)
// ---------------------------------------------------------------------------
__global__ void cam_gram(const _Float16* __restrict__ fh, float* __restrict__ ce,
                         int Bn, int Ci, int N) {
  int bid = blockIdx.x;
  int b  = bid / 16, r = bid % 16;
  int m0 = (r / 4) * 16, d0 = (r % 4) * 16;
  const int lane = threadIdx.x & (WAVE - 1);
  const int half = lane >> 4, lm = lane & 15;
  const _Float16* arow = fh + ((size_t)b * Ci + m0 + lm) * N;
  const _Float16* brow = fh + ((size_t)b * Ci + d0 + lm) * N;
  v8f acc = {};
  for (int k0 = 0; k0 < N; k0 += 32)
    acc = wmma_f16(load_a(arow + k0, half), load_b(brow + k0, half), acc);
#pragma unroll
  for (int rr = 0; rr < 8; ++rr) {
    int m = rr + 8 * half;
    ce[((size_t)b * Ci + m0 + m) * Ci + d0 + lm] = acc[rr];
  }
}

__global__ void cam_softmax(const float* __restrict__ ce, _Float16* __restrict__ cattn,
                            int Ci) {
  int b = blockIdx.x, c = threadIdx.x;       // 64 threads, one row each
  const float* row = ce + ((size_t)b * Ci + c) * Ci;
  float mx = -3.0e38f, mn = 3.0e38f;
  for (int d = 0; d < Ci; ++d) { float v = row[d]; mx = fmaxf(mx, v); mn = fminf(mn, v); }
  float s = 0.f;
  for (int d = 0; d < Ci; ++d) s += __expf(mn - row[d]);   // (mx-row)-(mx-mn)
  float inv = 1.f / s;
  for (int d = 0; d < Ci; ++d)
    cattn[((size_t)b * Ci + c) * Ci + d] = (_Float16)(__expf(mn - row[d]) * inv);
}

// cam = cattn @ f; writes sc_feat = gamma*cam + feat1 into PADDED ch-last layout
__global__ void cam_apply(const _Float16* __restrict__ cattn, const _Float16* __restrict__ f1T,
                          const float* __restrict__ feat1, const float* __restrict__ gamma,
                          _Float16* __restrict__ scP, int Bn, int Ci, int N, int H, int W) {
  const int ntiles = N / 16;
  int bid = blockIdx.x;
  int b  = bid / (ntiles * 4);
  int r  = bid % (ntiles * 4);
  int c0 = (r / ntiles) * 16;
  int n0 = (r % ntiles) * 16;
  const int lane = threadIdx.x & (WAVE - 1);
  const int half = lane >> 4, lm = lane & 15;
  float g = gamma[0];
  const _Float16* arow = cattn + ((size_t)b * Ci + c0 + lm) * Ci;
  const _Float16* brow = f1T + ((size_t)b * N + n0 + lm) * Ci;
  v8f acc = {};
  for (int k0 = 0; k0 < Ci; k0 += 32)
    acc = wmma_f16(load_a(arow + k0, half), load_b(brow + k0, half), acc);
  const int Wp = W + 2, Np = (H + 2) * Wp;
  const int n = n0 + lm, h = n / W, w = n % W;
  v8h hv;
#pragma unroll
  for (int rr = 0; rr < 8; ++rr) {
    int c = c0 + 8 * half + rr;
    float fv = feat1[((size_t)b * Ci + c) * N + n];
    hv[rr] = (_Float16)(g * acc[rr] + fv);
  }
  *(v8h*)&scP[((size_t)b * Np + (h + 1) * Wp + (w + 1)) * Ci + c0 + 8 * half] = hv;
}

// feat_sum = a + b (ch-first f32) -> channel-last (unpadded) f16 for conv8
__global__ void sum_to_chlast(const float* __restrict__ a, const float* __restrict__ b,
                              _Float16* __restrict__ outT, int C, int N, int total) {
  int i = blockIdx.x * blockDim.x + threadIdx.x;
  if (i >= total) return;
  int n = i % N, c = (i / N) % C, bb = i / (C * N);
  outT[((size_t)bb * N + n) * C + c] = (_Float16)(a[i] + b[i]);
}

// ---------------------------------------------------------------------------
// conv8: 1x1 64->256 GEMM + bias, M2-blocked (B fragment shared by 2 co-tiles)
// ---------------------------------------------------------------------------
__global__ void conv1x1_wmma(const _Float16* __restrict__ fT, const _Float16* __restrict__ wh,
                             const float* __restrict__ bias, float* __restrict__ out,
                             int Bn, int Cin, int Co, int N) {
  const int ntiles = N / 16, mtiles = Co / 32;
  int bid = blockIdx.x;
  int b  = bid / (ntiles * mtiles);
  int r  = bid % (ntiles * mtiles);
  int m0 = (r / ntiles) * 32;
  int n0 = (r % ntiles) * 16;
  const int lane = threadIdx.x & (WAVE - 1);
  const int half = lane >> 4, lm = lane & 15;
  const _Float16* arow = wh + (size_t)(m0 + lm) * Cin;
  const _Float16* brow = fT + ((size_t)b * N + n0 + lm) * Cin;
  v8f acc0 = {}, acc1 = {};
  for (int k0 = 0; k0 < Cin; k0 += 32) {
    v16h bf = load_b(brow + k0, half);
    acc0 = wmma_f16(load_a(arow + k0, half), bf, acc0);
    acc1 = wmma_f16(load_a(arow + (size_t)16 * Cin + k0, half), bf, acc1);
  }
#pragma unroll
  for (int rr = 0; rr < 8; ++rr) {
    int m = m0 + rr + 8 * half;
    out[((size_t)b * Co + m) * N + n0 + lm]      = acc0[rr] + bias[m];
    out[((size_t)b * Co + m + 16) * N + n0 + lm] = acc1[rr] + bias[m + 16];
  }
}

// ---------------------------------------------------------------------------
// Host launcher
// ---------------------------------------------------------------------------
static constexpr size_t alignUp(size_t x) { return (x + 255) & ~(size_t)255; }

extern "C" void kernel_launch(void* const* d_in, const int* in_sizes, int n_in,
                              void* d_out, int out_size, void* d_ws, size_t ws_size,
                              hipStream_t stream) {
  const int B = 4, Cin = 256, Ci = 64, Cq = 8, Co = 256, H = 64, W = 64;
  const int N = H * W;                       // 4096
  const int BN = B * N;                      // 16384
  const int Np = (H + 2) * (W + 2);          // 4356 (padded spatial)

  const float* x     = (const float*)d_in[0];
  const float* w1    = (const float*)d_in[1];
  const float* bn_g  = (const float*)d_in[2];
  const float* bn_b  = (const float*)d_in[3];
  const float* wq    = (const float*)d_in[4];
  const float* bq    = (const float*)d_in[5];
  const float* wk    = (const float*)d_in[6];
  const float* bk    = (const float*)d_in[7];
  const float* wv    = (const float*)d_in[8];
  const float* bv    = (const float*)d_in[9];
  const float* g_pam = (const float*)d_in[10];
  const float* g_cam = (const float*)d_in[11];
  const float* w2    = (const float*)d_in[12];
  const float* w8    = (const float*)d_in[13];
  const float* b8    = (const float*)d_in[14];
  float* out = (float*)d_out;

  char* ws = (char*)d_ws;
  size_t off = 0;
  auto alloc = [&](size_t bytes) { char* p = ws + off; off += alignUp(bytes); return p; };
  _Float16* xP     = (_Float16*)alloc((size_t)B * Np * Cin * 2); // padded ch-last
  _Float16* wt1    = (_Float16*)alloc((size_t)9 * Ci * Cin * 2); // [tap][Co][Cin]
  _Float16* wt2    = (_Float16*)alloc((size_t)9 * Ci * Ci * 2);
  _Float16* w8h    = (_Float16*)alloc((size_t)Co * Ci * 2);
  float*    y1     = (float*)   alloc((size_t)B * Ci * N * 4);
  float*    feat1  = (float*)   alloc((size_t)B * Ci * N * 4);   // ch-first f32
  _Float16* feat1h = (_Float16*)alloc((size_t)B * Ci * N * 2);   // ch-first f16 (gram)
  _Float16* f1T    = (_Float16*)alloc((size_t)B * N * Ci * 2);   // ch-last f16
  float*    scl    = (float*)   alloc(Ci * 4);
  float*    shf    = (float*)   alloc(Ci * 4);
  _Float16* qT     = (_Float16*)alloc((size_t)B * N * 32 * 2);
  _Float16* kT     = (_Float16*)alloc((size_t)B * N * 32 * 2);
  _Float16* vh     = (_Float16*)alloc((size_t)B * Ci * N * 2);   // ch-first
  _Float16* saP    = (_Float16*)alloc((size_t)B * Np * Ci * 2);  // padded ch-last
  _Float16* scP    = (_Float16*)alloc((size_t)B * Np * Ci * 2);  // padded ch-last
  float*    ce     = (float*)   alloc((size_t)B * Ci * Ci * 4);
  _Float16* cattn  = (_Float16*)alloc((size_t)B * Ci * Ci * 2);
  float*    y2a    = (float*)   alloc((size_t)B * Ci * N * 4);
  float*    y2b    = (float*)   alloc((size_t)B * Ci * N * 4);
  _Float16* fsumT  = (_Float16*)alloc((size_t)B * N * Ci * 2);
  (void)ws_size;

  const int T = 256;
  auto blk = [&](int n) { return (n + T - 1) / T; };

  // --- layout conversions (halo of conv2 inputs zeroed every call) ---
  pad_chlast_f16<<<blk(B * Np * Cin), T, 0, stream>>>(x, xP, B, Cin, H, W);
  zero_border<<<blk(B * Np * Ci), T, 0, stream>>>(saP, B, Ci, H, W);
  zero_border<<<blk(B * Np * Ci), T, 0, stream>>>(scP, B, Ci, H, W);
  repack_w3x3<<<blk(Ci * Cin * 9), T, 0, stream>>>(w1, wt1, Ci, Cin);
  repack_w3x3<<<blk(Ci * Ci * 9), T, 0, stream>>>(w2, wt2, Ci, Ci);
  cvt_f32_f16<<<blk(Co * Ci), T, 0, stream>>>(w8, w8h, Co * Ci);

  // --- conv1 + BN + ReLU -> feat1 (f32 + ch-first f16 + ch-last f16) ---
  conv3x3_wmma<<<B * (N / 16) * (Ci / 32), WAVE, 0, stream>>>(xP, wt1, y1, B, Cin, Ci, H, W);
  bn_stats<<<Ci, T, 0, stream>>>(y1, bn_g, bn_b, scl, shf, B, Ci, N);
  bn_relu_apply<<<blk(B * Ci * N), T, 0, stream>>>(y1, scl, shf, feat1, feat1h, f1T,
                                                   Ci, N, B * Ci * N);

  // --- PAM ---
  qkv_kernel<<<blk(BN), T, 0, stream>>>(feat1, wq, bq, wk, bk, wv, bv, qT, kT, vh, B, Ci, Cq, N);
  pam_flash<<<B * (N / 16), WAVE, 0, stream>>>(qT, kT, vh, feat1, g_pam, saP, B, Ci, N, H, W);

  // --- branch 1: conv2(sa_feat) + BN + ReLU ---
  conv3x3_wmma<<<B * (N / 16) * (Ci / 32), WAVE, 0, stream>>>(saP, wt2, y2a, B, Ci, Ci, H, W);
  bn_stats<<<Ci, T, 0, stream>>>(y2a, bn_g, bn_b, scl, shf, B, Ci, N);
  bn_relu_apply<<<blk(B * Ci * N), T, 0, stream>>>(y2a, scl, shf, y2a,
                                                   (_Float16*)nullptr, (_Float16*)nullptr,
                                                   Ci, N, B * Ci * N);

  // --- CAM ---
  cam_gram<<<B * 16, WAVE, 0, stream>>>(feat1h, ce, B, Ci, N);
  cam_softmax<<<B, Ci, 0, stream>>>(ce, cattn, Ci);
  cam_apply<<<B * (N / 16) * (Ci / 16), WAVE, 0, stream>>>(cattn, f1T, feat1, g_cam, scP,
                                                           B, Ci, N, H, W);

  // --- branch 2: conv2(sc_feat) + BN + ReLU ---
  conv3x3_wmma<<<B * (N / 16) * (Ci / 32), WAVE, 0, stream>>>(scP, wt2, y2b, B, Ci, Ci, H, W);
  bn_stats<<<Ci, T, 0, stream>>>(y2b, bn_g, bn_b, scl, shf, B, Ci, N);
  bn_relu_apply<<<blk(B * Ci * N), T, 0, stream>>>(y2b, scl, shf, y2b,
                                                   (_Float16*)nullptr, (_Float16*)nullptr,
                                                   Ci, N, B * Ci * N);

  // --- feat_sum -> conv8 ---
  sum_to_chlast<<<blk(B * Ci * N), T, 0, stream>>>(y2a, y2b, fsumT, Ci, N, B * Ci * N);
  conv1x1_wmma<<<B * (N / 16) * (Co / 32), WAVE, 0, stream>>>(fsumT, w8h, b8, out, B, Ci, Co, N);
}
